// MambaBlock_28089086116621
// MI455X (gfx1250) — matrix operand order repaired and checked by reading, not compiled, not run on hardware
//
#include <hip/hip_runtime.h>

// ---------------------------------------------------------------------------
// Types for CDNA5 WMMA
// ---------------------------------------------------------------------------
typedef __attribute__((ext_vector_type(16))) __bf16 v16bf;
typedef __attribute__((ext_vector_type(8)))  float  v8f;

static __device__ __forceinline__ __bf16 f2bf(float f) {
    unsigned u = __builtin_bit_cast(unsigned, f);
    unsigned r = (u + 0x7FFFu + ((u >> 16) & 1u)) >> 16;   // round-to-nearest-even
    unsigned short hs = (unsigned short)r;
    return __builtin_bit_cast(__bf16, hs);
}

static __device__ __forceinline__ float siluf(float v) {
    return v / (1.f + __expf(-v));
}

static __device__ __forceinline__ v8f wmma_bf16(v16bf a, v16bf b, v8f c) {
    return __builtin_amdgcn_wmma_f32_16x16x32_bf16(false, a, false, b,
                                                   (short)0, c, false, false);
}

// wave-relative LDS byte offset of a __shared__ object (AS(3) offset = low 32
// bits of the generic address on amdgcn)
static __device__ __forceinline__ unsigned lds_off(const void* p) {
    return (unsigned)(uintptr_t)p;
}

// CDNA5 async copy: cache/memory -> LDS, 16B per lane, tracked by ASYNCcnt
static __device__ __forceinline__ void async_copy_b128(unsigned lds, const void* gptr) {
    asm volatile("global_load_async_to_lds_b128 %0, %1, off"
                 :: "v"(lds), "v"(gptr) : "memory");
}
static __device__ __forceinline__ void wait_async0() {
    asm volatile("s_wait_asynccnt 0x0" ::: "memory");
}

// ---------------------------------------------------------------------------
// Problem constants
// ---------------------------------------------------------------------------
#define D_MODEL 1024
#define D_STATE 16
#define D_INNER 2048
#define BATCH   2
#define SEQ     2048
#define ROWS    (BATCH * SEQ)          // 4096 token rows
#define LDT     40                     // padded LDS row stride (bf16), 80B

// ---------------------------------------------------------------------------
// f32 -> bf16 cast (weights)
// ---------------------------------------------------------------------------
__global__ __launch_bounds__(256)
void cast_f32_bf16(const float* __restrict__ src, __bf16* __restrict__ dst, int n) {
    int i = blockIdx.x * 256 + threadIdx.x;
    if (i < n) dst[i] = f2bf(src[i]);
}

// ---------------------------------------------------------------------------
// LayerNorm over last dim (1024), emit bf16 normalized activations
// ---------------------------------------------------------------------------
__global__ __launch_bounds__(256)
void layernorm_bf16(const float* __restrict__ x, const float* __restrict__ w,
                    const float* __restrict__ b, __bf16* __restrict__ out) {
    const int row = blockIdx.x;
    const float* xr = x + (size_t)row * D_MODEL;
    float v[4], s = 0.f, ss = 0.f;
    for (int i = 0; i < 4; ++i) {
        v[i] = xr[threadIdx.x + i * 256];
        s += v[i]; ss += v[i] * v[i];
    }
    for (int off = 16; off; off >>= 1) { s += __shfl_xor(s, off); ss += __shfl_xor(ss, off); }
    __shared__ float rs[8], rss[8];
    if ((threadIdx.x & 31) == 0) { rs[threadIdx.x >> 5] = s; rss[threadIdx.x >> 5] = ss; }
    __syncthreads();
    float S = 0.f, SS = 0.f;
    for (int i = 0; i < 8; ++i) { S += rs[i]; SS += rss[i]; }
    const float mu  = S * (1.f / D_MODEL);
    const float var = SS * (1.f / D_MODEL) - mu * mu;
    const float inv = rsqrtf(var + 1e-5f);
    for (int i = 0; i < 4; ++i) {
        int col = threadIdx.x + i * 256;
        out[(size_t)row * D_MODEL + col] = f2bf((v[i] - mu) * inv * w[col] + b[col]);
    }
}

// ---------------------------------------------------------------------------
// Epilogue helper for the GEMMs
// MODE 0: plain f32 store
// MODE 1: softplus(acc + aux[col])          (dt path)
// MODE 2: aux[row*N+col] + acc              (residual output path)
// ---------------------------------------------------------------------------
template <int MODE>
static __device__ __forceinline__ void epi_store(float* __restrict__ C,
                                                 const float* __restrict__ aux,
                                                 int N, int r, int c, float v) {
    const size_t i = (size_t)r * N + c;
    if (MODE == 0) {
        C[i] = v;
    } else if (MODE == 1) {
        const float a = v + aux[c];
        C[i] = (a > 20.f) ? a : log1pf(__expf(a));
    } else {
        C[i] = aux[i] + v;
    }
}

// one K=32 compute stage: 8 ds_load_b128 fragments -> 4 WMMAs
static __device__ __forceinline__ void mma_step(const __bf16 (*__restrict__ Asp)[LDT],
                                                const __bf16 (*__restrict__ Bsp)[LDT],
                                                int wm, int wn, int lm, int g,
                                                v8f& a00, v8f& a01, v8f& a10, v8f& a11) {
    // A fragment: lane (g,m): K = {g*8..g*8+7} U {16+g*8..16+g*8+7}
    // B fragment: lane (g,n): K = {g*8..g*8+7} U {16+g*8..16+g*8+7} of column n
    union { v16bf v; uint4 q[2]; } fa0, fa1, fb0, fb1;
    fa0.q[0] = *(const uint4*)&Asp[wm      + lm][g * 8];
    fa0.q[1] = *(const uint4*)&Asp[wm      + lm][16 + g * 8];
    fa1.q[0] = *(const uint4*)&Asp[wm + 16 + lm][g * 8];
    fa1.q[1] = *(const uint4*)&Asp[wm + 16 + lm][16 + g * 8];
    fb0.q[0] = *(const uint4*)&Bsp[wn      + lm][g * 8];
    fb0.q[1] = *(const uint4*)&Bsp[wn      + lm][16 + g * 8];
    fb1.q[0] = *(const uint4*)&Bsp[wn + 16 + lm][g * 8];
    fb1.q[1] = *(const uint4*)&Bsp[wn + 16 + lm][16 + g * 8];
    a00 = wmma_bf16(fa0.v, fb0.v, a00);
    a01 = wmma_bf16(fa0.v, fb1.v, a01);
    a10 = wmma_bf16(fa1.v, fb0.v, a10);
    a11 = wmma_bf16(fa1.v, fb1.v, a11);
}

// ---------------------------------------------------------------------------
// WMMA bf16 GEMM:  C[M,N] = A[M,K] * Bw[N,K]^T   (weights stored row-major [N,K])
// Block tile 128x64, 256 threads = 8 waves, each wave -> 32x32 output
// (4 accumulators).  K staged in 32-chunks through double-buffered LDS filled
// by GLOBAL_LOAD_ASYNC_TO_LDS_B128; last stage peeled so the steady-state
// loop prefetches unconditionally with pure pointer increments.
// ---------------------------------------------------------------------------
template <int MODE>
__global__ __launch_bounds__(256)
void gemm_wmma_bf16(const __bf16* __restrict__ A, const __bf16* __restrict__ Bw,
                    float* __restrict__ Cout, const float* __restrict__ aux,
                    int M, int N, int K) {
    __shared__ __bf16 As[2][128][LDT];
    __shared__ __bf16 Bs[2][64][LDT];

    const int tid  = threadIdx.x;
    const int bm   = blockIdx.y * 128;
    const int bn   = blockIdx.x * 64;
    const int wave = tid >> 5, lane = tid & 31;
    const int wm   = (wave >> 1) * 32;      // 0,32,64,96
    const int wn   = (wave & 1) * 32;       // 0,32
    const int lm   = lane & 15, g = lane >> 4;

    // staging assignment: 16B chunks; A = 512 chunks (2/thread), B = 256 (1/thread)
    const int ar0 = tid >> 2;               // A rows 0..63
    const int ar1 = ar0 + 64;               // A rows 64..127
    const int ac  = (tid & 3) * 8;          // bf16 offset in row
    const __bf16* gA0 = A  + (size_t)(bm + ar0) * K + ac;
    const __bf16* gA1 = A  + (size_t)(bm + ar1) * K + ac;
    const __bf16* gB  = Bw + (size_t)(bn + ar0) * K + ac;

    unsigned ldsA0[2], ldsA1[2], ldsB[2];
    for (int p = 0; p < 2; ++p) {
        ldsA0[p] = lds_off(&As[p][ar0][ac]);
        ldsA1[p] = lds_off(&As[p][ar1][ac]);
        ldsB [p] = lds_off(&Bs[p][ar0][ac]);
    }

    // prologue: stage 0 into buffer 0
    async_copy_b128(ldsA0[0], gA0);
    async_copy_b128(ldsA1[0], gA1);
    async_copy_b128(ldsB [0], gB );
    gA0 += 32; gA1 += 32; gB += 32;
    wait_async0();
    __syncthreads();

    v8f acc00 = {}, acc01 = {}, acc10 = {}, acc11 = {};
    int p = 0;
    const int nstage = K >> 5;
    for (int s = 1; s < nstage; ++s) {      // steady state: unconditional prefetch
        async_copy_b128(ldsA0[p ^ 1], gA0);
        async_copy_b128(ldsA1[p ^ 1], gA1);
        async_copy_b128(ldsB [p ^ 1], gB );
        gA0 += 32; gA1 += 32; gB += 32;

        mma_step(As[p], Bs[p], wm, wn, lm, g, acc00, acc01, acc10, acc11);

        wait_async0();                      // next-stage tile landed in LDS
        __syncthreads();
        p ^= 1;
    }
    // peeled final stage: no prefetch, no wait
    mma_step(As[p], Bs[p], wm, wn, lm, g, acc00, acc01, acc10, acc11);

    // D layout: VGPR j -> M = g*8 + j ; col = lane&15 within each 16-wide tile
    for (int j = 0; j < 8; ++j) {
        const int r0 = bm + wm + g * 8 + j;
        const int r1 = r0 + 16;
        const int c0 = bn + wn + lm;
        const int c1 = c0 + 16;
        epi_store<MODE>(Cout, aux, N, r0, c0, acc00[j]);
        epi_store<MODE>(Cout, aux, N, r0, c1, acc01[j]);
        epi_store<MODE>(Cout, aux, N, r1, c0, acc10[j]);
        epi_store<MODE>(Cout, aux, N, r1, c1, acc11[j]);
    }
}

// ---------------------------------------------------------------------------
// Causal depthwise conv (K=4) + SiLU over x_part = xz[:, 0:2048]
// ---------------------------------------------------------------------------
__global__ __launch_bounds__(256)
void conv_silu(const float* __restrict__ xz, const float* __restrict__ cw,
               const float* __restrict__ cb, float* __restrict__ xcf,
               __bf16* __restrict__ xcb) {
    const int idx = blockIdx.x * 256 + threadIdx.x;     // ROWS*D_INNER
    const int c = idx & (D_INNER - 1);
    const int r = idx >> 11;                            // token row
    const int l = r & (SEQ - 1);                        // position inside batch
    float acc = cb[c];
    #pragma unroll
    for (int k = 0; k < 4; ++k) {
        const int lk = l + k - 3;
        if (lk >= 0) acc += xz[(size_t)(r + k - 3) * (2 * D_INNER) + c] * cw[c * 4 + k];
    }
    const float s = siluf(acc);
    xcf[idx] = s;
    xcb[idx] = f2bf(s);
}

// ---------------------------------------------------------------------------
// B/C projections: [ROWS, D_STATE] = x_conv[ROWS, D_INNER] * W^T, W = [16, 2048]
// ---------------------------------------------------------------------------
__global__ __launch_bounds__(32)
void bc_proj(const float* __restrict__ xc, const float* __restrict__ WB,
             const float* __restrict__ WC, float* __restrict__ Bm,
             float* __restrict__ Cm) {
    const int row = blockIdx.x;
    const int n   = blockIdx.y;
    const int lane = threadIdx.x;
    const float* xr = xc + (size_t)row * D_INNER;
    const float* wb = WB + (size_t)n * D_INNER;
    const float* wc = WC + (size_t)n * D_INNER;
    float ab = 0.f, ac = 0.f;
    for (int i = lane; i < D_INNER; i += 32) {
        const float xv = xr[i];
        ab += xv * wb[i];
        ac += xv * wc[i];
    }
    for (int off = 16; off; off >>= 1) { ab += __shfl_xor(ab, off); ac += __shfl_xor(ac, off); }
    if (lane == 0) { Bm[row * D_STATE + n] = ab; Cm[row * D_STATE + n] = ac; }
}

// ---------------------------------------------------------------------------
// Selective scan + skip (x_conv*D) + SiLU(z) gate, emitting bf16 for out-GEMM.
// 16 states per channel live in consecutive lanes; 16-lane butterfly -> y_t.
// ---------------------------------------------------------------------------
__global__ __launch_bounds__(256)
void scan_gate(const float* __restrict__ dt, const float* __restrict__ xcf,
               const float* __restrict__ Bm, const float* __restrict__ Cm,
               const float* __restrict__ A_log, const float* __restrict__ Dp,
               const float* __restrict__ xz, __bf16* __restrict__ ygb) {
    const int b  = blockIdx.x >> 7;                 // 128 channel-groups per batch
    const int cg = blockIdx.x & 127;
    const int c  = cg * 16 + (threadIdx.x >> 4);
    const int n  = threadIdx.x & 15;
    const float a  = -__expf(A_log[c * D_STATE + n]);
    const float Dv = Dp[c];
    float h = 0.f;
    const size_t base = (size_t)b * SEQ;
    for (int t = 0; t < SEQ; ++t) {
        const size_t row = base + t;
        const float dtv = dt [row * D_INNER + c];
        const float xv  = xcf[row * D_INNER + c];
        const float Bv  = Bm [row * D_STATE + n];
        const float Cv  = Cm [row * D_STATE + n];
        h = h * __expf(a * dtv) + xv * Bv * dtv;
        float y = h * Cv;
        y += __shfl_xor(y, 8); y += __shfl_xor(y, 4);
        y += __shfl_xor(y, 2); y += __shfl_xor(y, 1);
        if (n == 0) {
            const float z = xz[row * (2 * D_INNER) + D_INNER + c];
            ygb[row * D_INNER + c] = f2bf((y + xv * Dv) * siluf(z));
        }
    }
}

// ---------------------------------------------------------------------------
// Launch
// ---------------------------------------------------------------------------
extern "C" void kernel_launch(void* const* d_in, const int* in_sizes, int n_in,
                              void* d_out, int out_size, void* d_ws, size_t ws_size,
                              hipStream_t stream) {
    (void)in_sizes; (void)n_in; (void)out_size; (void)ws_size;
    const float* x      = (const float*)d_in[0];
    const float* norm_w = (const float*)d_in[1];
    const float* norm_b = (const float*)d_in[2];
    const float* W_in   = (const float*)d_in[3];
    const float* conv_w = (const float*)d_in[4];
    const float* conv_b = (const float*)d_in[5];
    const float* W_dt   = (const float*)d_in[6];
    const float* b_dt   = (const float*)d_in[7];
    const float* W_B    = (const float*)d_in[8];
    const float* W_C    = (const float*)d_in[9];
    const float* Dp     = (const float*)d_in[10];
    const float* A_log  = (const float*)d_in[11];
    const float* W_out  = (const float*)d_in[12];
    float* out = (float*)d_out;

    char* ws = (char*)d_ws;
    size_t off = 0;
    auto alloc = [&](size_t bytes) -> void* {
        void* p = ws + off;
        off += (bytes + 255) & ~(size_t)255;
        return p;
    };
    __bf16* xnb   = (__bf16*)alloc((size_t)ROWS * D_MODEL * 2);       // ln(x) bf16
    __bf16* Winb  = (__bf16*)alloc((size_t)2 * D_INNER * D_MODEL * 2);
    __bf16* Wdtb  = (__bf16*)alloc((size_t)D_INNER * D_INNER * 2);
    __bf16* Woutb = (__bf16*)alloc((size_t)D_MODEL * D_INNER * 2);
    float*  xzf   = (float*)alloc((size_t)ROWS * 2 * D_INNER * 4);    // xz f32
    float*  xcf   = (float*)alloc((size_t)ROWS * D_INNER * 4);        // silu(conv) f32
    __bf16* xcb   = (__bf16*)alloc((size_t)ROWS * D_INNER * 2);       // silu(conv) bf16
    float*  dtf   = (float*)alloc((size_t)ROWS * D_INNER * 4);        // softplus dt
    float*  Bmf   = (float*)alloc((size_t)ROWS * D_STATE * 4);
    float*  Cmf   = (float*)alloc((size_t)ROWS * D_STATE * 4);
    __bf16* ygb   = (__bf16*)alloc((size_t)ROWS * D_INNER * 2);       // gated y bf16

    // weight casts
    cast_f32_bf16<<<(2 * D_INNER * D_MODEL) / 256, 256, 0, stream>>>(W_in, Winb, 2 * D_INNER * D_MODEL);
    cast_f32_bf16<<<(D_INNER * D_INNER) / 256, 256, 0, stream>>>(W_dt, Wdtb, D_INNER * D_INNER);
    cast_f32_bf16<<<(D_MODEL * D_INNER) / 256, 256, 0, stream>>>(W_out, Woutb, D_MODEL * D_INNER);

    // layernorm -> bf16
    layernorm_bf16<<<ROWS, 256, 0, stream>>>(x, norm_w, norm_b, xnb);

    // xz = xn * W_in^T   [4096 x 4096], K=1024
    gemm_wmma_bf16<0><<<dim3((2 * D_INNER) / 64, ROWS / 128), 256, 0, stream>>>(
        xnb, Winb, xzf, nullptr, ROWS, 2 * D_INNER, D_MODEL);

    // causal conv + silu
    conv_silu<<<(ROWS * D_INNER) / 256, 256, 0, stream>>>(xzf, conv_w, conv_b, xcf, xcb);

    // dt = softplus(xc * W_dt^T + b_dt)   [4096 x 2048], K=2048
    gemm_wmma_bf16<1><<<dim3(D_INNER / 64, ROWS / 128), 256, 0, stream>>>(
        xcb, Wdtb, dtf, b_dt, ROWS, D_INNER, D_INNER);

    // B / C projections [4096 x 16]
    bc_proj<<<dim3(ROWS, D_STATE), 32, 0, stream>>>(xcf, W_B, W_C, Bmf, Cmf);

    // selective scan + gate -> bf16
    scan_gate<<<BATCH * (D_INNER / 16), 256, 0, stream>>>(
        dtf, xcf, Bmf, Cmf, A_log, Dp, xzf, ygb);

    // out = residual + y * W_out^T   [4096 x 1024], K=2048
    gemm_wmma_bf16<2><<<dim3(D_MODEL / 64, ROWS / 128), 256, 0, stream>>>(
        ygb, Woutb, out, x, ROWS, D_MODEL, D_INNER);
}
